// NetVLAD_53266184405649
// MI455X (gfx1250) — matrix operand order repaired and checked by reading, compiled, not run on hardware
//
#include <hip/hip_runtime.h>

// ---------------------------------------------------------------------------
// NetVLAD for MI455X (gfx1250, wave32, WMMA).
// N=64, C=512, H=W=32 (S=1024), K=64.  bf16 WMMA 16x16x32, f32 accumulate.
// Single HBM pass over x: per-pixel L2 norm fused into the main kernel.
// ---------------------------------------------------------------------------

#define N_DIM 64
#define C_DIM 512
#define S_DIM 1024
#define K_DIM 64
#define ST    64          // s-tile width per workgroup
#define NT    (S_DIM/ST)  // 16 s-tiles per image

typedef __attribute__((ext_vector_type(4)))  unsigned int u32x4;
typedef __attribute__((ext_vector_type(16))) __bf16       v16bf;
typedef __attribute__((ext_vector_type(8)))  float        v8f;
typedef __attribute__((ext_vector_type(4)))  float        f32x4;

union FragAB { u32x4 u[2]; v16bf v; };
union Acc    { v8f v; float f[8]; f32x4 q[2]; };

__device__ inline unsigned short f2bf(float f) {
    unsigned int u = __float_as_uint(f);
    unsigned int r = u + 0x7FFFu + ((u >> 16) & 1u);   // round-to-nearest-even
    return (unsigned short)(r >> 16);
}
__device__ inline float bf2f(unsigned short h) {
    return __uint_as_float(((unsigned int)h) << 16);
}
// Load one 16-half bf16 fragment as two 16-byte LDS reads (ds_read_b128 x2).
__device__ inline v16bf frag_ld(const unsigned short* p0, const unsigned short* p1) {
    FragAB f;
    f.u[0] = *(const u32x4*)p0;
    f.u[1] = *(const u32x4*)p1;
    return f.v;
}

// ---------------------------------------------------------------------------
// Kernel 0: zero the accumulation workspace (vlad partials, asum, sumsq).
// ---------------------------------------------------------------------------
__global__ void k0_zero(float* __restrict__ p, int count) {
    int i = blockIdx.x * 256 + threadIdx.x;
    if (i < count) p[i] = 0.0f;
}

// ---------------------------------------------------------------------------
// Kernel 1 (main): fused pixel-norm -> logits GEMM -> softmax(K) ->
// aggregation GEMM, all on one HBM pass over x.  bf16 WMMA 16x16x32.
// One workgroup (8 waves) per (n, s-tile of 64 columns).
// LDS layout (dynamic, 221184 B):
//   Wl [64][512]  bf16  conv_w, c-contiguous          (65536 B)
//   Xc [512][64]  bf16  x-tile, s-contiguous          (65536 B)
//   Xs [64][512]  bf16  x-tile, c-contiguous          (65536 B)
//   Lg [64][64]   f32   norm-reduce scratch, then logits (16384 B)
//   Al [64][64]   bf16  softmax weights a[k][s]       ( 8192 B)
// ---------------------------------------------------------------------------
__global__ __launch_bounds__(256) void k2_main(const float* __restrict__ x,
                                               const float* __restrict__ conv_w,
                                               float* __restrict__ vlad,
                                               float* __restrict__ asum) {
    extern __shared__ char smem[];
    unsigned short* Wl = (unsigned short*)smem;                 // [64][512]
    unsigned short* Xc = Wl + K_DIM * C_DIM;                    // [512][64]
    unsigned short* Xs = Xc + C_DIM * ST;                       // [64][512]
    float*          Lg = (float*)(Xs + ST * C_DIM);             // [64][64]
    unsigned short* Al = (unsigned short*)(Lg + ST * K_DIM);    // [64][64]

    const int tid = threadIdx.x;
    const int n   = blockIdx.x >> 4;         // NT = 16 s-tiles per image
    const int s0  = (blockIdx.x & 15) * ST;
    const int w   = tid >> 5;                // wave id 0..7
    const int lo  = tid & 15;                // lane % 16
    const int hi  = (tid >> 4) & 1;          // lane / 16 within wave32

    // ---- stage conv_w as bf16 ------------------------------------------------
    for (int i = tid; i < K_DIM * C_DIM; i += 256)
        Wl[i] = f2bf(conv_w[i]);

    // ---- single-pass load of x tile + per-pixel L2 norm ----------------------
    // Thread owns column j = tid&63 and channels c = cb + 4*m (m = 0..127),
    // holding its 128 f32 values in VGPRs while norms are reduced via LDS.
    {
        const float* xn = x + (size_t)n * C_DIM * S_DIM + s0;
        const int j  = tid & 63;
        const int cb = tid >> 6;             // 0..3
        float reg[128];
        float ss = 0.0f;
#pragma unroll
        for (int m = 0; m < 128; ++m) {
            if ((m & 15) == 0)               // global_prefetch_b8 hint
                __builtin_prefetch(xn + (size_t)(cb + 4 * (m + 16)) * S_DIM + j, 0, 0);
            float v = xn[(size_t)(cb + 4 * m) * S_DIM + j];
            reg[m] = v;
            ss = fmaf(v, v, ss);
        }
        Lg[cb * ST + j] = ss;                // 4 partials per column
        __syncthreads();
        float tot = Lg[j] + Lg[ST + j] + Lg[2 * ST + j] + Lg[3 * ST + j];
        float inv = 1.0f / fmaxf(sqrtf(tot), 1e-12f);
#pragma unroll
        for (int m = 0; m < 128; ++m) {
            int c = cb + 4 * m;
            unsigned short b = f2bf(reg[m] * inv);
            Xc[c * ST + j]    = b;           // s-contiguous (B-frags of GEMM 2)
            Xs[j * C_DIM + c] = b;           // c-contiguous (B-frags of GEMM 1)
        }
    }
    __syncthreads();                         // Xc/Xs ready; Lg free for logits

    // ---- GEMM 1: logits[k][s] = sum_c W[k][c] * X[c][s] ----------------------
    // 16 output tiles (kt 0..3 x st 0..3), 2 per wave, inner dim C in steps of 32.
    for (int t = w; t < 16; t += 8) {
        int kt = t >> 2, st = t & 3;
        Acc acc; acc.v = {};
        const unsigned short* Wrow = Wl + (kt * 16 + lo) * C_DIM;  // A: row = k
        const unsigned short* Xrow = Xs + (st * 16 + lo) * C_DIM;  // B: col = s
        for (int ci = 0; ci < C_DIM; ci += 32) {
            // A 16x32 bf16: halves 0-7 -> c0+8*hi+{0..7}, halves 8-15 -> +16
            v16bf a = frag_ld(Wrow + ci + hi * 8, Wrow + ci + hi * 8 + 16);
            // B 32x16 bf16: halves 0-7 -> c0+8*hi+{0..7}, halves 8-15 -> +16
            v16bf b = frag_ld(Xrow + ci + hi * 8, Xrow + ci + hi * 8 + 16);
            acc.v = __builtin_amdgcn_wmma_f32_16x16x32_bf16(
                false, a, false, b, (short)0, acc.v, false, false);
        }
        // D 16x16 f32: lane (lo,hi), VGPR j -> (M = 8*hi + j, N = lo)
        float* dst = Lg + (st * 16 + lo) * K_DIM + kt * 16 + hi * 8;
        *(f32x4*)dst       = acc.q[0];
        *(f32x4*)(dst + 4) = acc.q[1];
    }
    __syncthreads();

    // ---- softmax over K for each s column (threads 0..63, wave-uniform) ------
    if (tid < ST) {
        float* col = Lg + tid * K_DIM;
        float m = col[0];
        for (int k = 1; k < K_DIM; ++k) m = fmaxf(m, col[k]);
        float ssum = 0.0f;
        for (int k = 0; k < K_DIM; ++k) ssum += __expf(col[k] - m);
        float r = 1.0f / ssum;
        for (int k = 0; k < K_DIM; ++k)
            Al[k * ST + tid] = f2bf(__expf(col[k] - m) * r);
    }
    __syncthreads();

    // ---- partial asum[k] = sum_s a[k][s] (threads 0..63, wave-uniform) -------
    if (tid < K_DIM) {
        const unsigned short* row = Al + tid * ST;
        float s = 0.0f;
        for (int j = 0; j < ST; ++j) s += bf2f(row[j]);
        atomicAdd(&asum[n * K_DIM + tid], s);
    }

    // ---- GEMM 2: vlad[k][c] += sum_s a[k][s] * X[c][s] -----------------------
    // 128 output tiles (kt 0..3 x ct 0..31), 16 per wave, inner dim ST in 2 steps.
    float* vbase = vlad + (size_t)n * K_DIM * C_DIM;
    for (int i = 0; i < 16; ++i) {
        int t  = w * 16 + i;
        int kt = t & 3, ct = t >> 2;
        Acc acc; acc.v = {};
        const unsigned short* Ar = Al + (kt * 16 + lo) * ST;   // A: row = k
        const unsigned short* Xr = Xc + (ct * 16 + lo) * ST;   // B: col = c
        for (int si = 0; si < ST; si += 32) {
            v16bf a = frag_ld(Ar + si + hi * 8, Ar + si + hi * 8 + 16);
            v16bf b = frag_ld(Xr + si + hi * 8, Xr + si + hi * 8 + 16);
            acc.v = __builtin_amdgcn_wmma_f32_16x16x32_bf16(
                false, a, false, b, (short)0, acc.v, false, false);
        }
        int kbase = kt * 16 + hi * 8;
        int ccol  = ct * 16 + lo;
#pragma unroll
        for (int jj = 0; jj < 8; ++jj)
            atomicAdd(&vbase[(size_t)(kbase + jj) * C_DIM + ccol], acc.f[jj]);
    }
}

// ---------------------------------------------------------------------------
// Kernel 3: v = vlad - asum*centroid; intra-normalize over C; accumulate
// per-image sum of squares of the normalized vectors.
// One workgroup per (n,k); 256 threads, 2 channels each.
// ---------------------------------------------------------------------------
__global__ __launch_bounds__(256) void k3_intra(const float* __restrict__ vlad,
                                                const float* __restrict__ asum,
                                                const float* __restrict__ centroids,
                                                float* __restrict__ out,
                                                float* __restrict__ sumsq) {
    __shared__ float red[256];
    int b   = blockIdx.x;            // n*64 + k
    int n   = b >> 6;
    int k   = b & 63;
    int tid = threadIdx.x;

    float a = asum[b];
    const float* vrow = vlad + (size_t)b * C_DIM;
    const float* crow = centroids + (size_t)k * C_DIM;
    float v0 = vrow[tid]       - a * crow[tid];
    float v1 = vrow[tid + 256] - a * crow[tid + 256];
    red[tid] = v0 * v0 + v1 * v1;
    __syncthreads();
    for (int off = 128; off > 0; off >>= 1) {
        if (tid < off) red[tid] += red[tid + off];
        __syncthreads();
    }
    float total = red[0];
    float inv   = 1.0f / fmaxf(sqrtf(total), 1e-12f);
    float* orow = out + (size_t)b * C_DIM;
    orow[tid]       = v0 * inv;
    orow[tid + 256] = v1 * inv;
    if (tid == 0) atomicAdd(&sumsq[n], total * inv * inv);
}

// ---------------------------------------------------------------------------
// Kernel 4: global L2 normalization per image.
// ---------------------------------------------------------------------------
__global__ void k4_global(float* __restrict__ out, const float* __restrict__ sumsq) {
    int i = blockIdx.x * 256 + threadIdx.x;     // 0 .. N*K*C-1 (exact multiple)
    int n = i >> 15;                            // K*C = 32768
    out[i] *= 1.0f / fmaxf(sqrtf(sumsq[n]), 1e-12f);
}

// ---------------------------------------------------------------------------
extern "C" void kernel_launch(void* const* d_in, const int* in_sizes, int n_in,
                              void* d_out, int out_size, void* d_ws, size_t ws_size,
                              hipStream_t stream) {
    (void)in_sizes; (void)n_in; (void)out_size; (void)ws_size;
    const float* x         = (const float*)d_in[0];
    const float* conv_w    = (const float*)d_in[1];
    const float* centroids = (const float*)d_in[2];
    float* out = (float*)d_out;

    // Workspace layout (floats):
    float* wsf   = (float*)d_ws;
    float* vlad  = wsf;                                    // N*K*C = 2097152
    float* asum  = vlad + (size_t)N_DIM * K_DIM * C_DIM;   // N*K   = 4096
    float* sumsq = asum + N_DIM * K_DIM;                   // N     = 64

    const int zcount = N_DIM * K_DIM * C_DIM + N_DIM * K_DIM + N_DIM;
    k0_zero<<<(zcount + 255) / 256, 256, 0, stream>>>(vlad, zcount);

    const size_t smem = (size_t)(K_DIM * C_DIM * 2      // Wl
                               + C_DIM * ST * 2         // Xc
                               + ST * C_DIM * 2         // Xs
                               + ST * K_DIM * 4         // Lg
                               + K_DIM * ST * 2);       // Al  = 221184 B
    k2_main<<<N_DIM * NT, 256, smem, stream>>>(x, conv_w, vlad, asum);

    k3_intra<<<N_DIM * K_DIM, 256, 0, stream>>>(vlad, asum, centroids, out, sumsq);

    k4_global<<<(N_DIM * K_DIM * C_DIM) / 256, 256, 0, stream>>>(out, sumsq);
}
